// AtomToResidue_79791902425331
// MI455X (gfx1250) — matrix operand order repaired and checked by reading, not compiled, not run on hardware
//
#include <hip/hip_runtime.h>
#include <hip/hip_bf16.h>

#define B_    2
#define T_    8
#define A_    2048
#define E_    16384
#define F_    64
#define ED_   32
#define R_    128
#define NRES  256
#define BT_   (B_*T_)
#define KC_   288                 // fused comb K: 128 (src) + 128 (dst) + 32 (edge_attr)
#define VALID_BIT 0x40000000
#define OUT_EDGE_ELEMS ((size_t)BT_*NRES*NRES*R_)   // 134,217,728
#define OUT_NODE_ELEMS ((size_t)BT_*NRES*R_)        // 524,288

typedef __attribute__((ext_vector_type(16))) __bf16 v16bf;
typedef __attribute__((ext_vector_type(8)))  float  v8f;
typedef __attribute__((ext_vector_type(4)))  float  f32x4;   // clang vector for NT stores

union BF16x16 { unsigned u[8]; v16bf v; };

__device__ __forceinline__ unsigned pack2_bf16(float a, float b) {
    unsigned ua = __float_as_uint(a), ub = __float_as_uint(b);
    ua += 0x7FFFu + ((ua >> 16) & 1u);            // RNE
    ub += 0x7FFFu + ((ub >> 16) & 1u);
    return (ua >> 16) | (ub & 0xFFFF0000u);
}
__device__ __forceinline__ unsigned short f2bf(float a) {
    unsigned u = __float_as_uint(a);
    return (unsigned short)((u + 0x7FFFu + ((u >> 16) & 1u)) >> 16);
}
// monotone float<->uint encoding for atomicMax-based float max
__device__ __forceinline__ unsigned enc_f32(float f) {
    unsigned u = __float_as_uint(f);
    return (u & 0x80000000u) ? ~u : (u | 0x80000000u);
}
__device__ __forceinline__ float dec_f32(unsigned e) {
    unsigned u = (e & 0x80000000u) ? (e ^ 0x80000000u) : ~e;
    return __uint_as_float(u);
}
__device__ __forceinline__ void load8f_to_bf16(const float* p, unsigned* o) {
    float4 x = *reinterpret_cast<const float4*>(p);
    float4 y = *reinterpret_cast<const float4*>(p + 4);
    o[0] = pack2_bf16(x.x, x.y); o[1] = pack2_bf16(x.z, x.w);
    o[2] = pack2_bf16(y.x, y.y); o[3] = pack2_bf16(y.z, y.w);
}

// ---------------- init ----------------
// Output is 537MB >> 192MB L2: stream zeros with non-temporal stores so the
// weight panel / node_proj gather table stay cache-resident.
__global__ void zero_kernel(f32x4* p, long n4) {
    long i = (long)blockIdx.x * blockDim.x + threadIdx.x;
    long stride = (long)gridDim.x * blockDim.x;
    f32x4 z = (f32x4){0.f, 0.f, 0.f, 0.f};
    for (; i < n4; i += stride) __builtin_nontemporal_store(z, &p[i]);
}
__global__ void init_ws_kernel(unsigned* agg_enc, int* counts, int* has_edge) {
    int i = blockIdx.x * blockDim.x + threadIdx.x;
    if (i < BT_ * NRES * F_) agg_enc[i] = 0u;   // 0 == minimal encoded value
    if (i < BT_ * NRES)      counts[i] = 0;
    if (i < BT_)             has_edge[i] = 0;
}

// ---------------- segment max + counts + CA mask ----------------
// 4 threads per (bt,atom), 16 features each: shorter serialized atomic chains.
__global__ void atom_agg_kernel(const float* __restrict__ af, const int* __restrict__ residx,
                                unsigned* agg_enc, int* counts, int* ca_mask) {
    int idx = blockIdx.x * blockDim.x + threadIdx.x;
    if (idx >= BT_ * A_ * 4) return;
    int chunk = idx & 3;
    int aidx = idx >> 2;                        // (bt, atom)
    int bt = aidx / A_, a = aidx % A_, b = bt / T_;
    int res = residx[b * A_ + a];
    const float* f = af + (size_t)aidx * F_;
    if (chunk == 0) {
        ca_mask[aidx] = (f[1] > 0.5f) ? 1 : 0;  // CA_IDX = 1
        atomicAdd(&counts[bt * NRES + res], 1);
    }
    unsigned* row = agg_enc + (size_t)(bt * NRES + res) * F_ + chunk * 16;
    const float* fc = f + chunk * 16;
    #pragma unroll
    for (int k = 0; k < 16; ++k) atomicMax(&row[k], enc_f32(fc[k]));
}

// ---------------- per-edge meta ----------------
__global__ void edge_prep_kernel(const int* __restrict__ ei, const int* __restrict__ residx,
                                 const int* __restrict__ ca_mask, int2* meta, int* has_edge) {
    int idx = blockIdx.x * blockDim.x + threadIdx.x;
    if (idx >= BT_ * E_) return;
    int bt = idx / E_, e = idx % E_, b = bt / T_;
    int s = ei[((size_t)bt * 2 + 0) * E_ + e];
    int d = ei[((size_t)bt * 2 + 1) * E_ + e];
    int v = ca_mask[bt * A_ + s] & ca_mask[bt * A_ + d];
    int sres = residx[b * A_ + s], dres = residx[b * A_ + d];
    meta[idx] = make_int2(sres | (v ? VALID_BIT : 0), dres);
    if (v) atomicOr(&has_edge[bt], 1);
}

// ---------------- weight packing: bf16 [N][K] panels + fused W_edge@Wc_e + bias ----------------
__global__ void prep_weights_kernel(const float* __restrict__ W_atom, const float* __restrict__ W_edge,
                                    const float* __restrict__ W_c, const float* __restrict__ b_edge,
                                    const float* __restrict__ b_c,
                                    unsigned short* Bpk_comb, unsigned short* Bpk_atom, float* b_total) {
    int t = threadIdx.x;  // 256 threads
    for (int idx = t; idx < R_ * 256; idx += 256) {          // W_c[0:256] transposed
        int n = idx >> 8, k = idx & 255;
        Bpk_comb[n * KC_ + k] = f2bf(W_c[(size_t)k * R_ + n]);
    }
    for (int idx = t; idx < R_ * ED_; idx += 256) {          // W_ec = W_edge @ Wc_e
        int n = idx >> 5, kk = idx & 31;
        float s = 0.f;
        for (int j = 0; j < R_; ++j) s += W_edge[kk * R_ + j] * W_c[(size_t)(256 + j) * R_ + n];
        Bpk_comb[n * KC_ + 256 + kk] = f2bf(s);
    }
    for (int idx = t; idx < R_ * F_; idx += 256) {           // W_atom transposed
        int n = idx >> 6, k = idx & 63;
        Bpk_atom[n * F_ + k] = f2bf(W_atom[(size_t)k * R_ + n]);
    }
    if (t < R_) {                                            // b_total = b_c + b_edge @ Wc_e
        float s = b_c[t];
        for (int j = 0; j < R_; ++j) s += b_edge[j] * W_c[(size_t)(256 + j) * R_ + t];
        b_total[t] = s;
    }
}

__device__ __forceinline__ void load_bfrag(const unsigned short* bp, BF16x16& bfr) {
    uint4 b0 = *reinterpret_cast<const uint4*>(bp);
    uint4 b1 = *reinterpret_cast<const uint4*>(bp + 16);
    bfr.u[0] = b0.x; bfr.u[1] = b0.y; bfr.u[2] = b0.z; bfr.u[3] = b0.w;
    bfr.u[4] = b1.x; bfr.u[5] = b1.y; bfr.u[6] = b1.z; bfr.u[7] = b1.w;
}

// ---------------- node_proj = masked_agg @ W_atom + b_atom   (WMMA bf16) ----------------
__global__ void __launch_bounds__(128) node_gemm_kernel(const unsigned* __restrict__ agg_enc,
                                                        const int* __restrict__ counts,
                                                        const unsigned short* __restrict__ Bpk,
                                                        const float* __restrict__ b_atom,
                                                        float* __restrict__ node_proj) {
    const int lane = threadIdx.x & 31, w = threadIdx.x >> 5;
    const int wtile = blockIdx.x * 4 + w;          // 256 tiles over 4096 rows
    const int mbase = wtile * 16;
    const int l15 = lane & 15, hi = lane >> 4;
    const int arow = mbase + l15;
    const unsigned* ap = agg_enc + (size_t)arow * F_;
    const bool have = counts[arow] > 0;

    v8f acc[8];
    #pragma unroll
    for (int nt = 0; nt < 8; ++nt) acc[nt] = (v8f){0, 0, 0, 0, 0, 0, 0, 0};

    #pragma unroll
    for (int kt = 0; kt < 2; ++kt) {
        BF16x16 afr;
        const int kc = kt * 32 + hi * 8;
        #pragma unroll
        for (int c = 0; c < 2; ++c) {
            uint4 u0 = *reinterpret_cast<const uint4*>(ap + kc + c * 16);
            uint4 u1 = *reinterpret_cast<const uint4*>(ap + kc + c * 16 + 4);
            float f0 = have ? dec_f32(u0.x) : 0.f, f1 = have ? dec_f32(u0.y) : 0.f;
            float f2 = have ? dec_f32(u0.z) : 0.f, f3 = have ? dec_f32(u0.w) : 0.f;
            float f4 = have ? dec_f32(u1.x) : 0.f, f5 = have ? dec_f32(u1.y) : 0.f;
            float f6 = have ? dec_f32(u1.z) : 0.f, f7 = have ? dec_f32(u1.w) : 0.f;
            afr.u[c * 4 + 0] = pack2_bf16(f0, f1); afr.u[c * 4 + 1] = pack2_bf16(f2, f3);
            afr.u[c * 4 + 2] = pack2_bf16(f4, f5); afr.u[c * 4 + 3] = pack2_bf16(f6, f7);
        }
        #pragma unroll
        for (int nt = 0; nt < 8; ++nt) {
            BF16x16 bfr;
            load_bfrag(Bpk + (size_t)(nt * 16 + l15) * F_ + kt * 32 + hi * 8, bfr);
            acc[nt] = __builtin_amdgcn_wmma_f32_16x16x32_bf16(false, afr.v, false, bfr.v,
                                                              (short)0, acc[nt], false, false);
        }
    }
    float biasv[8];
    #pragma unroll
    for (int nt = 0; nt < 8; ++nt) biasv[nt] = b_atom[nt * 16 + l15];
    #pragma unroll
    for (int i = 0; i < 8; ++i) {
        const size_t row = (size_t)(mbase + i + hi * 8) * R_;
        #pragma unroll
        for (int nt = 0; nt < 8; ++nt)
            node_proj[row + nt * 16 + l15] = acc[nt][i] + biasv[nt];
    }
}

// ---------------- big fused GEMM + ReLU + LayerNorm + symmetric scatter ----------------
__global__ void __launch_bounds__(128) edge_gemm_kernel(const float* __restrict__ edge_attr,
                                                        const int2* __restrict__ meta,
                                                        const float* __restrict__ node_proj,
                                                        const unsigned short* __restrict__ Bpk,
                                                        const float* __restrict__ b_total,
                                                        const float* __restrict__ gamma,
                                                        const float* __restrict__ beta,
                                                        float* __restrict__ out) {
    const int lane = threadIdx.x & 31, w = threadIdx.x >> 5;
    const int wtile = blockIdx.x * 4 + w;        // 16384 tiles = BT * E/16
    const int bt = wtile >> 10;
    const int mbase = (wtile & 1023) * 16;
    const int l15 = lane & 15, hi = lane >> 4;

    const int arow = mbase + l15;
    const int2 m = meta[(size_t)bt * E_ + arow];
    const int sres = m.x & 0xFFFF;
    const int dres = m.y;
    const float* srow = node_proj + ((size_t)bt * NRES + sres) * R_;
    const float* drow = node_proj + ((size_t)bt * NRES + dres) * R_;
    const float* erow = edge_attr + ((size_t)bt * E_ + arow) * ED_;

    v8f acc[8];
    #pragma unroll
    for (int nt = 0; nt < 8; ++nt) acc[nt] = (v8f){0, 0, 0, 0, 0, 0, 0, 0};

    #pragma unroll
    for (int kt = 0; kt < 9; ++kt) {
        BF16x16 afr;
        const int kc0 = kt * 32 + hi * 8;
        #pragma unroll
        for (int c = 0; c < 2; ++c) {
            const int kc = kc0 + c * 16;
            const float* p = (kc < 128) ? (srow + kc)
                            : (kc < 256) ? (drow + (kc - 128))
                                         : (erow + (kc - 256));
            load8f_to_bf16(p, &afr.u[c * 4]);
        }
        #pragma unroll
        for (int nt = 0; nt < 8; ++nt) {
            BF16x16 bfr;
            load_bfrag(Bpk + (size_t)(nt * 16 + l15) * KC_ + kt * 32 + hi * 8, bfr);
            acc[nt] = __builtin_amdgcn_wmma_f32_16x16x32_bf16(false, afr.v, false, bfr.v,
                                                              (short)0, acc[nt], false, false);
        }
    }

    float biasv[8], gv[8], bev[8];
    #pragma unroll
    for (int nt = 0; nt < 8; ++nt) {
        int n = nt * 16 + l15;
        biasv[nt] = b_total[n]; gv[nt] = gamma[n]; bev[nt] = beta[n];
    }
    #pragma unroll
    for (int nt = 0; nt < 8; ++nt)
        #pragma unroll
        for (int i = 0; i < 8; ++i) {
            float x = acc[nt][i] + biasv[nt];
            acc[nt][i] = x > 0.f ? x : 0.f;               // ReLU
        }
    // LayerNorm per row (row i + hi*8): 128 values across 8 nt x 16 lanes of this half
    #pragma unroll
    for (int i = 0; i < 8; ++i) {
        float s = 0.f, q = 0.f;
        #pragma unroll
        for (int nt = 0; nt < 8; ++nt) { float x = acc[nt][i]; s += x; q += x * x; }
        #pragma unroll
        for (int msk = 1; msk < 16; msk <<= 1) {
            s += __shfl_xor(s, msk, 32);
            q += __shfl_xor(q, msk, 32);
        }
        float mu = s * (1.f / R_);
        float var = q * (1.f / R_) - mu * mu;
        float rs = rsqrtf(var + 1e-5f);
        #pragma unroll
        for (int nt = 0; nt < 8; ++nt)
            acc[nt][i] = (acc[nt][i] - mu) * rs * gv[nt] + bev[nt];
    }
    // symmetric scatter of valid edges into dense residue-pair grid (streaming, NT)
    #pragma unroll
    for (int i = 0; i < 8; ++i) {
        const int row = mbase + i + hi * 8;
        const int2 mm = meta[(size_t)bt * E_ + row];
        if (mm.x & VALID_BIT) {
            const int s0 = mm.x & 0xFFFF, d0 = mm.y;
            const size_t p0 = (((size_t)bt * NRES + s0) * NRES + d0) * R_;
            const size_t p1 = (((size_t)bt * NRES + d0) * NRES + s0) * R_;
            #pragma unroll
            for (int nt = 0; nt < 8; ++nt) {
                const int n = nt * 16 + l15;
                const float v = acc[nt][i];
                __builtin_nontemporal_store(v, &out[p0 + n]);
                __builtin_nontemporal_store(v, &out[p1 + n]);
            }
        }
    }
}

// ---------------- masked node-feature output ----------------
__global__ void node_out_kernel(const float* __restrict__ node_proj, const int* __restrict__ counts,
                                const int* __restrict__ has_edge, float* __restrict__ out_node) {
    int idx = blockIdx.x * blockDim.x + threadIdx.x;
    if (idx >= BT_ * NRES * R_) return;
    int bt = idx / (NRES * R_);
    int res = (idx / R_) % NRES;
    bool msk = has_edge[bt] && (counts[bt * NRES + res] > 0);
    __builtin_nontemporal_store(msk ? node_proj[idx] : 0.f, &out_node[idx]);
}

extern "C" void kernel_launch(void* const* d_in, const int* in_sizes, int n_in,
                              void* d_out, int out_size, void* d_ws, size_t ws_size,
                              hipStream_t stream) {
    const float* atom_features   = (const float*)d_in[0];
    const int*   residue_indices = (const int*)  d_in[1];
    const int*   edge_index      = (const int*)  d_in[2];
    const float* edge_attr       = (const float*)d_in[3];
    const float* W_atom          = (const float*)d_in[4];
    const float* b_atom          = (const float*)d_in[5];
    const float* W_edge          = (const float*)d_in[6];
    const float* b_edge          = (const float*)d_in[7];
    const float* W_c             = (const float*)d_in[8];
    const float* b_c             = (const float*)d_in[9];
    const float* ln_gamma        = (const float*)d_in[10];
    const float* ln_beta         = (const float*)d_in[11];
    float* out = (float*)d_out;
    float* out_node = out + OUT_EDGE_ELEMS;

    char* ws = (char*)d_ws;
    size_t off = 0;
    auto take = [&](size_t bytes) { size_t o = off; off = (off + bytes + 255) & ~(size_t)255; return o; };
    unsigned*       agg_enc  = (unsigned*)      (ws + take((size_t)BT_ * NRES * F_ * 4));
    int*            counts   = (int*)           (ws + take((size_t)BT_ * NRES * 4));
    int*            ca_mask  = (int*)           (ws + take((size_t)BT_ * A_ * 4));
    int2*           meta     = (int2*)          (ws + take((size_t)BT_ * E_ * 8));
    int*            has_edge = (int*)           (ws + take((size_t)BT_ * 4));
    float*          node_proj= (float*)         (ws + take((size_t)BT_ * NRES * R_ * 4));
    unsigned short* Bpk_comb = (unsigned short*)(ws + take((size_t)R_ * KC_ * 2));
    unsigned short* Bpk_atom = (unsigned short*)(ws + take((size_t)R_ * F_ * 2));
    float*          b_total  = (float*)         (ws + take((size_t)R_ * 4));
    (void)in_sizes; (void)n_in; (void)out_size; (void)ws_size;

    const long n4 = (long)((OUT_EDGE_ELEMS + OUT_NODE_ELEMS) / 4);
    zero_kernel<<<8192, 256, 0, stream>>>((f32x4*)out, n4);
    init_ws_kernel<<<(BT_ * NRES * F_ + 255) / 256, 256, 0, stream>>>(agg_enc, counts, has_edge);
    prep_weights_kernel<<<1, 256, 0, stream>>>(W_atom, W_edge, W_c, b_edge, b_c,
                                               Bpk_comb, Bpk_atom, b_total);
    atom_agg_kernel<<<(BT_ * A_ * 4 + 255) / 256, 256, 0, stream>>>(atom_features, residue_indices,
                                                                    agg_enc, counts, ca_mask);
    edge_prep_kernel<<<(BT_ * E_ + 255) / 256, 256, 0, stream>>>(edge_index, residue_indices,
                                                                 ca_mask, meta, has_edge);
    node_gemm_kernel<<<(BT_ * NRES / 16) / 4, 128, 0, stream>>>(agg_enc, counts, Bpk_atom,
                                                                b_atom, node_proj);
    edge_gemm_kernel<<<(BT_ * E_ / 16) / 4, 128, 0, stream>>>(edge_attr, meta, node_proj,
                                                              Bpk_comb, b_total, ln_gamma,
                                                              ln_beta, out);
    node_out_kernel<<<(BT_ * NRES * R_ + 255) / 256, 256, 0, stream>>>(node_proj, counts,
                                                                       has_edge, out_node);
}
